// CoAtNetTransformerClassifier_31705448579164
// MI455X (gfx1250) — compile-verified, matrix operand order checked by reading, not run on hardware
//
// CoAtNet forward for MI455X (gfx1250), compile-only target.
//
// Canonical activation layout: NHWC bf16. All dense contractions -> bf16 WMMA
// GEMM (v_wmma_f32_16x16x32_bf16, f32 accumulate):
//   * 1x1 convs: activation IS the GEMM A matrix (direct, double-buffered
//     ASYNCcnt pipeline: issue tile k+1, s_wait_asynccnt 3, consume tile k)
//   * 3x3 convs: implicit im2col with K=(kh,kw,ci); Ci%8==0 makes each 8-elem
//     chunk a single 16B GLOBAL_LOAD_ASYNC_TO_LDS_B128
//   * attention/MLP/head: batched token GEMMs (grid.z = B*heads)
// Out-of-range M/N tiles use clamped (branchless) loads; garbage only reaches
// masked output rows/cols. K padding is zeroed at operand-build time.
//
// Input leaf order assumption: harness flattens setup_inputs() like a JAX
// pytree (dict keys sorted): params(62 leaves, nested sorted) then x.

#include <hip/hip_runtime.h>
#include <math.h>

typedef unsigned short u16;
typedef unsigned int   u32;
typedef __bf16 v16bf __attribute__((ext_vector_type(16)));
typedef float  v8f   __attribute__((ext_vector_type(8)));

#define RSQ_BN 0.99999500003749969f   // 1/sqrt(1+1e-5)

__device__ __forceinline__ u16 f2bf(float x) {
  u32 u = __float_as_uint(x);
  u32 r = u + 0x7FFFu + ((u >> 16) & 1u);   // round-to-nearest-even
  return (u16)(r >> 16);
}
__device__ __forceinline__ float bf2f(u16 v) { return __uint_as_float(((u32)v) << 16); }
__device__ __forceinline__ float siluf(float x) { return x / (1.f + __expf(-x)); }
__device__ __forceinline__ float geluf(float x) { return 0.5f * x * (1.f + erff(x * 0.70710678118f)); }
__device__ __forceinline__ int mini(int a, int b) { return a < b ? a : b; }

// ---- CDNA5 async global->LDS (ASYNCcnt) -----------------------------------
typedef __attribute__((address_space(3))) u16 as3_u16;

__device__ __forceinline__ u32 lds_off32(void* p) {
  return (u32)(uintptr_t)(as3_u16*)(u16*)p;   // flat -> LDS byte offset
}
__device__ __forceinline__ void async_g2l_b128(u32 ldsOff, const u16* g) {
  asm volatile("global_load_async_to_lds_b128 %0, %1, off"
               :: "v"(ldsOff), "v"((unsigned long long)(uintptr_t)g)
               : "memory");
}
__device__ __forceinline__ void wait_async0() {
  asm volatile("s_wait_asynccnt 0" ::: "memory");
}
__device__ __forceinline__ void wait_async3() {   // <=3 outstanding (one group)
  asm volatile("s_wait_asynccnt 3" ::: "memory");
}

// ---------------------------------------------------------------- GEMM params
struct GP {
  const u16* A;          // MODE 0: direct A (M,Kp) bf16 row-major
  const u16* Bw;         // weights (Nc,Kp) bf16 row-major
  float*     C;          // f32 output (or null)
  u16*       Cbf;        // bf16 output (or null)
  const float* Res;      // residual (row-major), optional
  const float* gs;       // bn gamma (epi 1/2)
  const float* bs;       // bn beta / bias (epi 1..4)
  const float* X;        // MODE 1: f32 NCHW source (stem0)
  const u16*  Xb;        // MODE 2: bf16 NHWC source
  long long sA, sB, sC;  // per-blockIdx.z strides (elements)
  int M, Nc, Kp, Kreal;
  int epi;               // 0 none, 1 bn+silu, 2 bn, 3 bias+gelu, 4 bias
  int Ci, Hin, Win, st, pad;        // conv params
  int Pout, PoutSh, Wout, WoutSh;   // Pout = 1<<PoutSh, Wout = 1<<WoutSh
  int CiSh;                          // log2(Ci/3) for MODE 2 (Ci = 3<<CiSh)
};

union Frag { uint4 q[2]; v16bf v; };

// 128(M) x 64(N) tile, Ktile=32, 256 threads = 8 waves (wave32),
// each wave: 32x32 = 2x2 wmma_f32_16x16x32_bf16 tiles.
// MODE 0: direct bf16 A, double-buffered async pipeline.
// MODE 1: NCHW f32 im2col (stem0, Ci=3), scalar gather.
// MODE 2: NHWC bf16 im2col, K=(kh,kw,ci), 16B async chunks.
template <int MODE>
__global__ __launch_bounds__(256) void k_gemm(GP p) {
  __shared__ __align__(16) u16 la[2][128 * 40];
  __shared__ __align__(16) u16 lb[2][64 * 40];
  const int tid  = threadIdx.x;
  const int wave = tid >> 5, lane = tid & 31;
  const int wm = wave & 3, wn = wave >> 2;
  const int m0 = blockIdx.x * 128, n0 = blockIdx.y * 64;
  const int z  = blockIdx.z;
  const u16* Bg = p.Bw + (size_t)z * (size_t)p.sB;

  v8f c[2][2] = {};
  const int hl = lane >> 4, l16 = lane & 15;

  // consume one staged k-tile: fragment loads + 4 WMMAs
  auto consume = [&](int buf) {
    Frag fa[2], fb[2];
#pragma unroll
    for (int t = 0; t < 2; t++) {
      int row = wm * 32 + t * 16 + l16;
      fa[t].q[0] = *(const uint4*)&la[buf][row * 40 + hl * 8];
      fa[t].q[1] = *(const uint4*)&la[buf][row * 40 + 16 + hl * 8];
      int col = wn * 32 + t * 16 + l16;
      fb[t].q[0] = *(const uint4*)&lb[buf][col * 40 + hl * 8];
      fb[t].q[1] = *(const uint4*)&lb[buf][col * 40 + 16 + hl * 8];
    }
#pragma unroll
    for (int ti = 0; ti < 2; ti++)
#pragma unroll
      for (int tj = 0; tj < 2; tj++)
        c[ti][tj] = __builtin_amdgcn_wmma_f32_16x16x32_bf16(
            false, fa[ti].v, false, fb[tj].v, (short)0, c[ti][tj], false, false);
  };
  // branchless B-tile issue (clamped row; garbage cols are store-masked)
  auto issueB = [&](int kt, int buf) {
    int r = tid >> 2, c8 = (tid & 3) * 8;
    int n = mini(n0 + r, p.Nc - 1);
    async_g2l_b128(lds_off32(&lb[buf][r * 40 + c8]),
                   Bg + (size_t)n * p.Kp + kt + c8);
  };

  if (MODE == 0) {
    // -------- double-buffered ASYNCcnt pipeline (3 asyncs per wave per tile)
    const u16* Ag = p.A + (size_t)z * (size_t)p.sA;
    auto issue = [&](int kt, int buf) {
#pragma unroll
      for (int i = 0; i < 2; i++) {
        int r = (tid >> 2) + i * 64;
        int c8 = (tid & 3) * 8;
        int m = mini(m0 + r, p.M - 1);
        async_g2l_b128(lds_off32(&la[buf][r * 40 + c8]),
                       Ag + (size_t)m * p.Kp + kt + c8);
      }
      issueB(kt, buf);
    };
    const int nsteps = p.Kp >> 5;
    issue(0, 0);
    for (int s = 0; s < nsteps; s++) {
      int cur = s & 1;
      bool more = (s + 1 < nsteps);
      if (more) issue((s + 1) << 5, cur ^ 1);
      if (more) wait_async3(); else wait_async0();
      __syncthreads();
      consume(cur);
      __syncthreads();
    }
  } else {
    // -------- single-buffered im2col paths
    for (int kt = 0; kt < p.Kp; kt += 32) {
      if (MODE == 1) {          // stem0: scalar NCHW f32 gather (tiny layer)
        const size_t plane = (size_t)p.Hin * p.Win;
#pragma unroll
        for (int i = 0; i < 2; i++) {
          int r = (tid >> 2) + i * 64;
          int c8 = (tid & 3) * 8;
          int m = m0 + r;
          union { u16 u[8]; uint4 q; } t;
#pragma unroll
          for (int e = 0; e < 8; e++) {
            int k = kt + c8 + e;
            float v = 0.f;
            if (m < p.M && k < p.Kreal) {
              int bb = m >> p.PoutSh, pp = m & (p.Pout - 1);
              int oy = pp >> p.WoutSh, ox = pp & (p.Wout - 1);
              int ci = k / 9; int rr = k - ci * 9;
              int kh = rr / 3, kw = rr - kh * 3;
              int iy = oy * p.st - p.pad + kh;
              int ix = ox * p.st - p.pad + kw;
              if (iy >= 0 && iy < p.Hin && ix >= 0 && ix < p.Win) {
                size_t idx = ((size_t)bb * p.Ci + ci) * plane + (size_t)iy * p.Win + ix;
                v = p.X[idx];
                if (e == 0) __builtin_prefetch((const void*)(p.X + idx + plane), 0, 1);
              }
            }
            t.u[e] = f2bf(v);
          }
          *(uint4*)&la[0][r * 40 + c8] = t.q;
        }
      } else {                  // MODE 2: NHWC bf16, 16B async chunk per tap
#pragma unroll
        for (int i = 0; i < 2; i++) {
          int r = (tid >> 2) + i * 64;
          int c8 = (tid & 3) * 8;
          int m = mini(m0 + r, p.M - 1);
          int k0 = kt + c8;
          u32 loff = lds_off32(&la[0][r * 40 + c8]);
          int t9  = (k0 >> p.CiSh) / 3;       // k0 / Ci, Ci = 3<<CiSh
          int ci0 = k0 - t9 * p.Ci;
          int kh = t9 / 3, kw = t9 - kh * 3;
          int bb = m >> p.PoutSh, pp = m & (p.Pout - 1);
          int oy = pp >> p.WoutSh, ox = pp & (p.Wout - 1);
          int iy = oy * p.st - p.pad + kh;
          int ix = ox * p.st - p.pad + kw;
          if (iy >= 0 && iy < p.Hin && ix >= 0 && ix < p.Win) {
            async_g2l_b128(loff,
                p.Xb + (((size_t)bb * p.Hin + iy) * p.Win + ix) * p.Ci + ci0);
          } else {              // real 3x3 halo: must be zeros
            uint4 z4 = {0u, 0u, 0u, 0u};
            *(uint4*)&la[0][r * 40 + c8] = z4;
          }
        }
      }
      issueB(kt, 0);
      wait_async0();
      __syncthreads();
      consume(0);
      __syncthreads();
    }
  }

  // ---- epilogue (row-major (M,Nc); coalesced along n)
  float* Cg = p.C ? p.C + (size_t)z * (size_t)p.sC : nullptr;
  u16*   Cb = p.Cbf ? p.Cbf + (size_t)z * (size_t)p.sC : nullptr;
  const float* Rg = p.Res ? p.Res + (size_t)z * (size_t)p.sC : nullptr;
#pragma unroll
  for (int ti = 0; ti < 2; ti++)
#pragma unroll
    for (int tj = 0; tj < 2; tj++)
#pragma unroll
      for (int v = 0; v < 8; v++) {
        int m = m0 + wm * 32 + ti * 16 + hl * 8 + v;
        int n = n0 + wn * 32 + tj * 16 + l16;
        if (m < p.M && n < p.Nc) {
          float x = c[ti][tj][v];
          if (p.epi == 1 || p.epi == 2)      x = x * (p.gs[n] * RSQ_BN) + p.bs[n];
          else if (p.epi == 3 || p.epi == 4) x += p.bs[n];
          if (p.epi == 1)      x = siluf(x);
          else if (p.epi == 3) x = geluf(x);
          size_t off = (size_t)m * p.Nc + n;
          if (Rg) x += Rg[off];
          if (Cb) Cb[off] = f2bf(x); else Cg[off] = x;
        }
      }
}

// ------------------------------------------------------------ helper kernels
#define GTID ((long long)blockIdx.x * blockDim.x + threadIdx.x)

__global__ void k_wconv(const float* s, u16* d, int N, int K, int Kp, long long tot) {
  long long i = GTID; if (i >= tot) return;
  int n = (int)(i / Kp), k = (int)(i % Kp);
  d[i] = f2bf(k < K ? s[(size_t)n * K + k] : 0.f);
}

// (Co, Ci, 3, 3) -> (Co, (kh*3+kw)*Ci + ci) bf16
__global__ void k_wconv9(const float* s, u16* d, int N, int Ci, long long tot) {
  long long i = GTID; if (i >= tot) return;
  int K = 9 * Ci;
  int n = (int)(i / K); int rr = (int)(i % K);
  int t9 = rr / Ci, ci = rr - t9 * Ci;
  d[i] = f2bf(s[(size_t)n * K + (size_t)ci * 9 + t9]);
}

__global__ void k_zero16(u16* p, long long tot) {
  long long i = GTID; if (i < tot) p[i] = 0;
}

__global__ void k_f2b(const float* s, u16* d, long long tot) {
  long long i = GTID; if (i < tot) d[i] = f2bf(s[i]);
}

// depthwise 3x3 + BN + SiLU on NHWC bf16 -> NHWC f32 (channel-fastest: coalesced)
__global__ void k_dwnhwc(const u16* X, const float* w, const float* g, const float* b,
                         float* Y, int B, int C, int H, int W, int stp, int Ho, int Wo) {
  long long i = GTID;
  long long tot = (long long)B * Ho * Wo * C; if (i >= tot) return;
  int c = (int)(i % C); long long t = i / C;
  int wo = (int)(t % Wo); t /= Wo;
  int ho = (int)(t % Ho); int bb = (int)(t / Ho);
  float acc = 0.f;
#pragma unroll
  for (int kh = 0; kh < 3; kh++)
#pragma unroll
    for (int kw = 0; kw < 3; kw++) {
      int iy = ho * stp - 1 + kh, ix = wo * stp - 1 + kw;
      if (iy >= 0 && iy < H && ix >= 0 && ix < W)
        acc += bf2f(X[(((size_t)bb * H + iy) * W + ix) * C + c]) * w[c * 9 + kh * 3 + kw];
    }
  acc = acc * (g[c] * RSQ_BN) + b[c];
  Y[i] = siluf(acc);
}

// mean over spatial for NHWC f32: s[b*C+c]
__global__ void k_pool_nhwc(const float* X, float* s, int B, int C, int P) {
  long long i = GTID; if (i >= (long long)B * C) return;
  int c = (int)(i % C); int bb = (int)(i / C);
  float a = 0.f;
  for (int p = 0; p < P; p++) a += X[((size_t)bb * P + p) * C + c];
  s[i] = a / (float)P;
}

__global__ void k_se1(const float* s, const float* w1, const float* b1, float* s1,
                      int B, int Hc, int Se) {
  long long i = GTID; if (i >= (long long)B * Se) return;
  int j = (int)(i % Se), bb = (int)(i / Se);
  float a = b1[j];
  for (int c = 0; c < Hc; c++) a += w1[(size_t)j * Hc + c] * s[(size_t)bb * Hc + c];
  s1[i] = siluf(a);
}

__global__ void k_se2(const float* s1, const float* w2, const float* b2, float* s2,
                      int B, int Hc, int Se) {
  long long i = GTID; if (i >= (long long)B * Hc) return;
  int c = (int)(i % Hc), bb = (int)(i / Hc);
  float a = b2[c];
  for (int j = 0; j < Se; j++) a += w2[(size_t)c * Se + j] * s1[(size_t)bb * Se + j];
  s2[i] = 1.f / (1.f + __expf(-a));
}

// NHWC f32 * per-(b,c) scale -> NHWC bf16
__global__ void k_scalecvt(const float* X, const float* s2, u16* Y, int B, int C, int P) {
  long long i = GTID; if (i >= (long long)B * P * C) return;
  int c = (int)(i % C);
  int bb = (int)(i / ((long long)P * C));
  Y[i] = f2bf(X[i] * s2[(size_t)bb * C + c]);
}

__global__ void k_ln(const float* T, const float* g, const float* b, u16* U, int M, int C) {
  long long i = GTID; if (i >= M) return;
  const float* p = T + (size_t)i * C;
  float m = 0.f;
  for (int c = 0; c < C; c++) m += p[c];
  m /= (float)C;
  float v = 0.f;
  for (int c = 0; c < C; c++) { float d = p[c] - m; v += d * d; }
  v /= (float)C;
  float inv = rsqrtf(v + 1e-5f);
  u16* o = U + (size_t)i * C;
  for (int c = 0; c < C; c++) o[c] = f2bf((p[c] - m) * inv * g[c] + b[c]);
}

__global__ void k_relbias(const float* rpb, float* bias, int H, int heads) {
  int N = H * H;
  long long i = GTID; if (i >= (long long)heads * N * N) return;
  int h = (int)(i / ((long long)N * N));
  long long r = i % ((long long)N * N);
  int ii = (int)(r / N), jj = (int)(r % N);
  int ih = ii / H, iw = ii % H, jh = jj / H, jw = jj % H;
  int dh = ih - jh + H - 1, dw = iw - jw + H - 1;
  int ri = dh * (2 * H - 1) + dw;
  bias[i] = rpb[(size_t)ri * heads + h];
}

__global__ void k_qkvrepack(const float* QKV, u16* Qp, u16* Kp, u16* Vp,
                            int B, int N, int heads, int hd) {
  long long i = GTID; if (i >= (long long)B * N * 3 * heads * hd) return;
  int C3 = 3 * heads * hd;
  int j = (int)(i % C3); long long r = i / C3;
  int n = (int)(r % N); int bb = (int)(r / N);
  float v = QKV[i];
  int wq = j / (heads * hd), rc = j % (heads * hd);
  int h = rc / hd, d = rc % hd;
  size_t bh = (size_t)bb * heads + h;
  if (wq == 0)      Qp[(bh * N + n) * 32 + d] = f2bf(v * 0.20412414523193154f); // hd^-0.5
  else if (wq == 1) Kp[(bh * N + n) * 32 + d] = f2bf(v);
  else              Vp[(bh * hd + d) * N + n] = f2bf(v);
}

__global__ void k_softmax(const float* S, const float* bias, u16* P, int Bh, int N, int heads) {
  long long i = GTID; if (i >= (long long)Bh * N) return;
  int bh = (int)(i / N), ii = (int)(i % N);
  int h = bh % heads;
  const float* sp = S + ((size_t)bh * N + ii) * N;
  const float* bp = bias + ((size_t)h * N + ii) * N;
  float mx = -1e30f;
  for (int j = 0; j < N; j++) { float v = sp[j] + bp[j]; mx = v > mx ? v : mx; }
  float sum = 0.f;
  for (int j = 0; j < N; j++) sum += __expf(sp[j] + bp[j] - mx);
  float inv = 1.f / sum;
  u16* op = P + ((size_t)bh * N + ii) * N;
  for (int j = 0; j < N; j++) op[j] = f2bf(__expf(sp[j] + bp[j] - mx) * inv);
}

__global__ void k_orepack(const float* O, u16* Obf, int B, int N, int heads, int hd) {
  int C = heads * hd;
  long long i = GTID; if (i >= (long long)B * N * C) return;
  int c = (int)(i % C); long long r = i / C;
  int n = (int)(r % N); int bb = (int)(r / N);
  int h = c / hd, d = c % hd;
  Obf[i] = f2bf(O[(((size_t)bb * heads + h) * N + n) * hd + d]);
}

// tokens (B,N,C) f32 -> pooled (B,C) bf16
__global__ void k_gap_tok(const float* T, u16* o, int B, int N, int C) {
  long long i = GTID; if (i >= (long long)B * C) return;
  int c = (int)(i % C); int bb = (int)(i / C);
  float a = 0.f;
  for (int n = 0; n < N; n++) a += T[((size_t)bb * N + n) * C + c];
  o[i] = f2bf(a / (float)N);
}

// -------------------------------------------------------------------- driver
static inline int ilog2i(int v) { int s = 0; while ((1 << s) < v) s++; return s; }

extern "C" void kernel_launch(void* const* d_in, const int* in_sizes, int n_in,
                              void* d_out, int out_size, void* d_ws, size_t ws_size,
                              hipStream_t stream) {
  (void)in_sizes; (void)n_in; (void)out_size; (void)ws_size;
  // leaf indices under JAX pytree (sorted-key) flattening
  enum { BLK1 = 0, BLK2 = 11, CONV1 = 22, CONV2 = 35, DOWN1 = 48, DOWN2 = 51,
         HEAD_B = 54, HEAD_W = 55, STEM0 = 56, STEM1 = 59, XIN = 62 };
  enum { FC1B = 0, FC1W, FC2B, FC2W, LN1B, LN1G, LN2B, LN2G, PROJW, QKVW, RPBI };
  enum { DWB = 0, DWG, DWW, EXB, EXG, EXW, PRB, PRG, PRW, SEB1, SEB2, SEW1, SEW2 };
  enum { CB = 0, CG, CW };
  auto F = [&](int i) { return (const float*)d_in[i]; };

  char* wsbase = (char*)d_ws;
  size_t off = 0;
  auto alloc = [&](size_t bytes) -> void* {
    size_t a = (off + 255) & ~(size_t)255;
    off = a + bytes;
    return (void*)(wsbase + a);
  };
  auto EL = [&](long long n) { return dim3((unsigned)((n + 255) / 256)); };

  // ---- bf16 weight conversion
  auto wconv = [&](const float* src, int N, int K) -> u16* {   // flat (N,Kp)
    int Kp = (K + 31) & ~31;
    u16* d = (u16*)alloc((size_t)N * Kp * 2);
    long long tot = (long long)N * Kp;
    k_wconv<<<EL(tot), 256, 0, stream>>>(src, d, N, K, Kp, tot);
    return d;
  };
  auto wconv9 = [&](const float* src, int N, int Ci) -> u16* { // (N, khkw*Ci)
    long long tot = (long long)N * 9 * Ci;
    u16* d = (u16*)alloc((size_t)tot * 2);
    k_wconv9<<<EL(tot), 256, 0, stream>>>(src, d, N, Ci, tot);
    return d;
  };

  u16* Wstem0 = wconv (F(STEM0 + CW), 96, 27);     // MODE 1 (ci-major)
  u16* Wstem1 = wconv9(F(STEM1 + CW), 96, 96);     // MODE 2 (tap-major)
  u16* Wc1ex  = wconv (F(CONV1 + EXW), 384, 96);
  u16* Wc1pr  = wconv (F(CONV1 + PRW), 96, 384);
  u16* Wc2ex  = wconv (F(CONV2 + EXW), 384, 96);
  u16* Wc2pr  = wconv (F(CONV2 + PRW), 192, 384);
  u16* Wd1    = wconv9(F(DOWN1 + CW), 192, 192);
  u16* Wd2    = wconv9(F(DOWN2 + CW), 192, 192);
  u16* Wqkv1  = wconv (F(BLK1 + QKVW), 576, 192);
  u16* Wpr1   = wconv (F(BLK1 + PROJW), 192, 192);
  u16* Wf11   = wconv (F(BLK1 + FC1W), 768, 192);
  u16* Wf21   = wconv (F(BLK1 + FC2W), 192, 768);
  u16* Wqkv2  = wconv (F(BLK2 + QKVW), 576, 192);
  u16* Wpr2   = wconv (F(BLK2 + PROJW), 192, 192);
  u16* Wf12   = wconv (F(BLK2 + FC1W), 768, 192);
  u16* Wf22   = wconv (F(BLK2 + FC2W), 192, 768);
  u16* Whead  = wconv (F(HEAD_W), 1000, 192);

  // ---- activation buffers (NHWC bf16 ping-pong) + attention buffers
  u16*   ACT1 = (u16*)alloc((size_t)32 * 16384 * 384 * 2);  // 402 MB
  u16*   ACT2 = (u16*)alloc((size_t)32 * 16384 * 384 * 2);  // 402 MB
  float* DWF  = (float*)alloc((size_t)32 * 4096 * 384 * 4); // 201 MB
  float* sPool = (float*)alloc((size_t)32 * 384 * 4);
  float* s1 = (float*)alloc((size_t)32 * 96 * 4);
  float* s2 = (float*)alloc((size_t)32 * 384 * 4);
  float* Tt  = (float*)alloc((size_t)8192 * 192 * 4);
  float* T2  = (float*)alloc((size_t)8192 * 192 * 4);
  float* T3  = (float*)alloc((size_t)8192 * 192 * 4);
  u16*   Ubf = (u16*)alloc((size_t)8192 * 192 * 2);
  float* QKV = (float*)alloc((size_t)8192 * 576 * 4);
  u16*   Qp  = (u16*)alloc((size_t)256 * 256 * 32 * 2);
  u16*   Kpb = (u16*)alloc((size_t)256 * 256 * 32 * 2);
  u16*   Vp  = (u16*)alloc((size_t)256 * 24 * 256 * 2);
  float* S   = (float*)alloc((size_t)256 * 256 * 256 * 4);
  u16*   Pbf = (u16*)alloc((size_t)256 * 256 * 256 * 2);
  float* Obuf = (float*)alloc((size_t)256 * 256 * 24 * 4);
  u16*   Obf  = (u16*)alloc((size_t)8192 * 192 * 2);
  float* BiasB = (float*)alloc((size_t)8 * 256 * 256 * 4);
  u16*   U2bf = (u16*)alloc((size_t)8192 * 768 * 2);
  u16*   Abf  = (u16*)alloc((size_t)8192 * 192 * 2);
  u16*   pooled = (u16*)alloc((size_t)32 * 192 * 2);

  // ---- generic GEMM launcher (all modes)
  auto gemm = [&](int mode, GP& p, int batches) {
    dim3 g3((p.M + 127) / 128, (p.Nc + 63) / 64, batches);
    if (mode == 0)      k_gemm<0><<<g3, 256, 0, stream>>>(p);
    else if (mode == 1) k_gemm<1><<<g3, 256, 0, stream>>>(p);
    else                k_gemm<2><<<g3, 256, 0, stream>>>(p);
  };

  // direct (1x1 conv / token) GEMM; NHWC/tokens row-major in & out
  auto mm = [&](const u16* A, const u16* Bw, int M, int Nc, int Kp, int batches,
                long long sA, long long sB, long long sC, int epi,
                const float* gs, const float* bs, const float* Res,
                float* C, u16* Cb) {
    GP p = {};
    p.A = A; p.Bw = Bw; p.C = C; p.Cbf = Cb; p.Res = Res; p.gs = gs; p.bs = bs;
    p.sA = sA; p.sB = sB; p.sC = sC;
    p.M = M; p.Nc = Nc; p.Kp = Kp; p.Kreal = Kp; p.epi = epi;
    gemm(0, p, batches);
  };

  // 3x3 conv over NHWC bf16 (MODE 2)
  auto conv3 = [&](const u16* Xb, int Ci, int H, int W, int stp,
                   const u16* Wb, int Co, const float* g, const float* b,
                   int epi, float* outF, u16* outB) {
    int Ho = (H + 2 - 3) / stp + 1, Wo = (W + 2 - 3) / stp + 1;
    GP p = {};
    p.Bw = Wb; p.C = outF; p.Cbf = outB; p.gs = g; p.bs = b; p.Xb = Xb;
    p.M = 32 * Ho * Wo; p.Nc = Co; p.Kp = 9 * Ci; p.Kreal = p.Kp; p.epi = epi;
    p.Ci = Ci; p.Hin = H; p.Win = W; p.st = stp; p.pad = 1;
    p.Pout = Ho * Wo; p.PoutSh = ilog2i(p.Pout);
    p.Wout = Wo; p.WoutSh = ilog2i(Wo);
    p.CiSh = ilog2i(Ci / 3);
    gemm(2, p, 1);
  };

  auto se = [&](const float* h, int Hc, int Se, int P,
                const float* w1, const float* b1, const float* w2, const float* b2) {
    k_pool_nhwc<<<EL((long long)32 * Hc), 256, 0, stream>>>(h, sPool, 32, Hc, P);
    k_se1<<<EL((long long)32 * Se), 256, 0, stream>>>(sPool, w1, b1, s1, 32, Hc, Se);
    k_se2<<<EL((long long)32 * Hc), 256, 0, stream>>>(s1, w2, b2, s2, 32, Hc, Se);
  };

  // ---- transformer block: input tokens already in Tt (B,N,C) f32; out -> T3
  auto block = [&](int Htok, int base, const u16* Wq, const u16* Wp,
                   const u16* Wf1, const u16* Wf2) {
    const int N = Htok * Htok, M = 32 * N, C = 192, heads = 8, hd = 24;
    const int Bh = 32 * heads;
    k_ln<<<EL(M), 256, 0, stream>>>(Tt, F(base + LN1G), F(base + LN1B), Ubf, M, C);
    mm(Ubf, Wq, M, 3 * C, C, 1, 0, 0, 0, 0, nullptr, nullptr, nullptr, QKV, nullptr);
    long long qk = (long long)Bh * N * 32;
    k_zero16<<<EL(qk), 256, 0, stream>>>(Qp, qk);
    k_zero16<<<EL(qk), 256, 0, stream>>>(Kpb, qk);
    long long tot = (long long)32 * N * 3 * C;
    k_qkvrepack<<<EL(tot), 256, 0, stream>>>(QKV, Qp, Kpb, Vp, 32, N, heads, hd);
    tot = (long long)heads * N * N;
    k_relbias<<<EL(tot), 256, 0, stream>>>(F(base + RPBI), BiasB, Htok, heads);
    mm(Qp, Kpb, N, N, 32, Bh, (long long)N * 32, (long long)N * 32, (long long)N * N,
       0, nullptr, nullptr, nullptr, S, nullptr);
    tot = (long long)Bh * N;
    k_softmax<<<EL(tot), 256, 0, stream>>>(S, BiasB, Pbf, Bh, N, heads);
    mm(Pbf, Vp, N, hd, N, Bh, (long long)N * N, (long long)hd * N, (long long)N * hd,
       0, nullptr, nullptr, nullptr, Obuf, nullptr);
    tot = (long long)M * C;
    k_orepack<<<EL(tot), 256, 0, stream>>>(Obuf, Obf, 32, N, heads, hd);
    mm(Obf, Wp, M, C, C, 1, 0, 0, 0, 0, nullptr, nullptr, Tt, T2, nullptr);
    k_ln<<<EL(M), 256, 0, stream>>>(T2, F(base + LN2G), F(base + LN2B), Ubf, M, C);
    mm(Ubf, Wf1, M, 4 * C, C, 1, 0, 0, 0, 3, nullptr, F(base + FC1B), nullptr, nullptr, U2bf);
    mm(U2bf, Wf2, M, C, 4 * C, 1, 0, 0, 0, 4, nullptr, F(base + FC2B), T2, T3, nullptr);
  };

  // =================================================================== graph
  // stem0: NCHW f32 x -> NHWC bf16 (MODE 1)
  {
    GP p = {};
    p.Bw = Wstem0; p.Cbf = ACT1; p.gs = F(STEM0 + CG); p.bs = F(STEM0 + CB);
    p.X = F(XIN);
    p.M = 32 * 16384; p.Nc = 96; p.Kp = 32; p.Kreal = 27; p.epi = 1;
    p.Ci = 3; p.Hin = 128; p.Win = 128; p.st = 1; p.pad = 1;
    p.Pout = 16384; p.PoutSh = 14; p.Wout = 128; p.WoutSh = 7;
    gemm(1, p, 1);
  }
  // stem1: 3x3 NHWC (96 -> 96) @128
  conv3(ACT1, 96, 128, 128, 1, Wstem1, 96, F(STEM1 + CG), F(STEM1 + CB), 1, nullptr, ACT2);
  // conv1 (MBConv 96->96, stride 2): expand 1x1 (direct A)
  mm(ACT2, Wc1ex, 32 * 16384, 384, 96, 1, 0, 0, 0, 1,
     F(CONV1 + EXG), F(CONV1 + EXB), nullptr, nullptr, ACT1);
  k_dwnhwc<<<EL((long long)32 * 64 * 64 * 384), 256, 0, stream>>>(
      ACT1, F(CONV1 + DWW), F(CONV1 + DWG), F(CONV1 + DWB), DWF, 32, 384, 128, 128, 2, 64, 64);
  se(DWF, 384, 96, 64 * 64, F(CONV1 + SEW1), F(CONV1 + SEB1), F(CONV1 + SEW2), F(CONV1 + SEB2));
  k_scalecvt<<<EL((long long)32 * 4096 * 384), 256, 0, stream>>>(DWF, s2, ACT2, 32, 384, 4096);
  mm(ACT2, Wc1pr, 32 * 4096, 96, 384, 1, 0, 0, 0, 2,
     F(CONV1 + PRG), F(CONV1 + PRB), nullptr, nullptr, ACT1);
  // conv2 (MBConv 96->192, stride 2)
  mm(ACT1, Wc2ex, 32 * 4096, 384, 96, 1, 0, 0, 0, 1,
     F(CONV2 + EXG), F(CONV2 + EXB), nullptr, nullptr, ACT2);
  k_dwnhwc<<<EL((long long)32 * 32 * 32 * 384), 256, 0, stream>>>(
      ACT2, F(CONV2 + DWW), F(CONV2 + DWG), F(CONV2 + DWB), DWF, 32, 384, 64, 64, 2, 32, 32);
  se(DWF, 384, 96, 32 * 32, F(CONV2 + SEW1), F(CONV2 + SEB1), F(CONV2 + SEW2), F(CONV2 + SEB2));
  k_scalecvt<<<EL((long long)32 * 1024 * 384), 256, 0, stream>>>(DWF, s2, ACT1, 32, 384, 1024);
  mm(ACT1, Wc2pr, 32 * 1024, 192, 384, 1, 0, 0, 0, 2,
     F(CONV2 + PRG), F(CONV2 + PRB), nullptr, nullptr, ACT2);
  // down1: 3x3 stride 2 (192->192) @32 -> tokens (B,256,192) f32 in Tt
  conv3(ACT2, 192, 32, 32, 2, Wd1, 192, F(DOWN1 + CG), F(DOWN1 + CB), 1, Tt, nullptr);
  block(16, BLK1, Wqkv1, Wpr1, Wf11, Wf21);
  // down2: 3x3 stride 2 @16 on block1 output (tokens are NHWC)
  k_f2b<<<EL((long long)8192 * 192), 256, 0, stream>>>(T3, Abf, (long long)8192 * 192);
  conv3(Abf, 192, 16, 16, 2, Wd2, 192, F(DOWN2 + CG), F(DOWN2 + CB), 1, Tt, nullptr);
  block(8, BLK2, Wqkv2, Wpr2, Wf12, Wf22);
  // global average pool + classifier head
  k_gap_tok<<<EL((long long)32 * 192), 256, 0, stream>>>(T3, pooled, 32, 64, 192);
  mm(pooled, Whead, 32, 1000, 192, 1, 0, 0, 0, 4,
     nullptr, F(HEAD_B), nullptr, (float*)d_out, nullptr);
}
